// MSARowAttentionWithPairBias_77017353552607
// MI455X (gfx1250) — matrix operand order repaired and checked by reading, not compiled, not run on hardware
//
#include <hip/hip_runtime.h>
#include <hip/hip_bf16.h>

// ---------------- problem constants ----------------
#define N_SEQ  256
#define N_RES  256
#define C_M    256
#define C_Z    128
#define C_H    32
#define N_HEAD 8
#define M_ROWS (N_SEQ * N_RES)   // 65536

typedef __attribute__((ext_vector_type(16))) _Float16 v16h;
typedef __attribute__((ext_vector_type(8)))  _Float16 v8h;
typedef __attribute__((ext_vector_type(8)))  float    v8f;
typedef __attribute__((ext_vector_type(4)))  float    v4f;

// ---------------- helpers ----------------
__device__ __forceinline__ v8f zero8() {
  v8f z;
  #pragma unroll
  for (int i = 0; i < 8; ++i) z[i] = 0.0f;
  return z;
}

__device__ __forceinline__ v16h make_v16(v8h lo, v8h hi) {
  v16h r;
  #pragma unroll
  for (int i = 0; i < 8; ++i) { r[i] = lo[i]; r[i + 8] = hi[i]; }
  return r;
}

// A-fragment (16x32 f16, MxK, row-major source, ld = elements per row).
// Lane L: row = L&15, half = L>>4. elems 0..7 -> K = half*8 + j,
// elems 8..15 -> K = 16 + half*8 + j  (ISA 7.12.2, 16-bit A 16x32).
__device__ __forceinline__ v16h load_frag_a(const _Float16* p, int ld) {
  const int lane = threadIdx.x & 31;
  const int r = lane & 15, hf = lane >> 4;
  const _Float16* row = p + (size_t)r * ld;
  v8h lo = *(const v8h*)(row + hf * 8);
  v8h hi = *(const v8h*)(row + 16 + hf * 8);
  return make_v16(lo, hi);
}

// B-fragment (32x16 f16, KxN) loaded from Bt stored as [N][K] row-major.
// Lane L: col n = L&15, half = L>>4, elems 0..7 -> K = half*8 + j,
// elems 8..15 -> K = 16 + half*8 + j (same order as A).
__device__ __forceinline__ v16h load_frag_b(const _Float16* p, int ld) {
  const int lane = threadIdx.x & 31;
  const int n = lane & 15, hf = lane >> 4;
  const _Float16* row = p + (size_t)n * ld;
  v8h lo = *(const v8h*)(row + hf * 8);
  v8h hi = *(const v8h*)(row + 16 + hf * 8);
  return make_v16(lo, hi);
}

// CDNA5 async memory->LDS copy (ASYNCcnt tracked, no VGPR round trip).
// Copies 32 bytes: b128 at +0 and b128 at +16 (INST_OFFSET applies to BOTH
// the LDS and the global address, ISA 15.18.3 op 98).
__device__ __forceinline__ void async_copy_32B(void* lds_dst, const void* gsrc) {
  unsigned lds_addr = (unsigned)(size_t)lds_dst;          // low 32 bits = LDS offset
  unsigned long long gaddr = (unsigned long long)(size_t)gsrc;
  asm volatile("global_load_async_to_lds_b128 %0, %1, off"
               :: "v"(lds_addr), "v"(gaddr) : "memory");
  asm volatile("global_load_async_to_lds_b128 %0, %1, off offset:16"
               :: "v"(lds_addr), "v"(gaddr) : "memory");
}

__device__ __forceinline__ void wait_async_all() {
  asm volatile("s_wait_asynccnt 0x0" ::: "memory");
}

__device__ __forceinline__ float block_reduce_sum(float v, float* red) {
  #pragma unroll
  for (int m = 16; m >= 1; m >>= 1) v += __shfl_xor(v, m, 32);
  const int wid = threadIdx.x >> 5, lane = threadIdx.x & 31;
  const int nw = blockDim.x >> 5;
  if (lane == 0) red[wid] = v;
  __syncthreads();
  if (threadIdx.x == 0) {
    float r = red[0];
    for (int i = 1; i < nw; ++i) r += red[i];
    red[0] = r;
  }
  __syncthreads();
  float r = red[0];
  __syncthreads();           // allow red[] reuse by caller
  return r;
}

// ---------------- kernel 1: weight f32 -> f16 transposed [N][K] ----------------
__global__ void prep_weights_kernel(const float* __restrict__ Wq, const float* __restrict__ Wk,
                                    const float* __restrict__ Wv, const float* __restrict__ Wg,
                                    const float* __restrict__ Wo,
                                    _Float16* __restrict__ wqT, _Float16* __restrict__ wkT,
                                    _Float16* __restrict__ wvT, _Float16* __restrict__ wgT,
                                    _Float16* __restrict__ woT) {
  const int k = blockIdx.x;      // source row (K)
  const int n = threadIdx.x;     // source col (N)
  const float* src; _Float16* dst;
  switch (blockIdx.y) {
    case 0:  src = Wq; dst = wqT; break;
    case 1:  src = Wk; dst = wkT; break;
    case 2:  src = Wv; dst = wvT; break;
    case 3:  src = Wg; dst = wgT; break;
    default: src = Wo; dst = woT; break;
  }
  dst[(size_t)n * 256 + k] = (_Float16)src[(size_t)k * 256 + n];
}

// ---------------- kernel 2: layernorm(m) -> f16 ----------------
__global__ __launch_bounds__(256)
void ln_m_kernel(const float* __restrict__ m, const float* __restrict__ w,
                 const float* __restrict__ b, _Float16* __restrict__ mn) {
  __shared__ float red[8];
  const int row = blockIdx.x;
  const int c = threadIdx.x;
  float x = m[(size_t)row * C_M + c];
  float mu = block_reduce_sum(x, red) * (1.0f / C_M);
  float d = x - mu;
  float var = block_reduce_sum(d * d, red) * (1.0f / C_M);
  float y = d * rsqrtf(var + 1e-5f) * w[c] + b[c];
  mn[(size_t)row * C_M + c] = (_Float16)y;
}

// ---------------- kernel 3: pair bias, written in WMMA C-fragment layout ------
// bias_frag[h][qt][kt][lane][e]  (f32), so the attention kernel adds bias with
// two contiguous b128 loads per 16x16 tile instead of 8 scattered b32 loads.
// C-fragment mapping: row-in-tile = e + 8*(lane>>4), col-in-tile = lane&15.
__global__ __launch_bounds__(128)
void pair_bias_kernel(const float* __restrict__ z, const float* __restrict__ w,
                      const float* __restrict__ b, const float* __restrict__ Wz,
                      float* __restrict__ bias_frag) {
  __shared__ float red[4];
  __shared__ float zn[C_Z];
  const int i = blockIdx.y, j = blockIdx.x;     // (qrow, kcol)
  const int c = threadIdx.x;
  float x = z[((size_t)i * N_RES + j) * C_Z + c];
  float mu = block_reduce_sum(x, red) * (1.0f / C_Z);
  float d = x - mu;
  float var = block_reduce_sum(d * d, red) * (1.0f / C_Z);
  zn[c] = d * rsqrtf(var + 1e-5f) * w[c] + b[c];
  __syncthreads();
  if (c < N_HEAD) {
    float acc = 0.0f;
    #pragma unroll 8
    for (int t = 0; t < C_Z; ++t) acc += zn[t] * Wz[(size_t)t * N_HEAD + c];
    const int qt = i >> 4, e = i & 7, hh = (i >> 3) & 1;
    const int kt = j >> 4, lane = hh * 16 + (j & 15);
    bias_frag[((((size_t)c * 16 + qt) * 16 + kt) * 32 + lane) * 8 + e] = acc;
  }
}

// ---------------- kernel 4: generic WMMA GEMM, M=65536 N=256 K=256 ----------------
// C = A(f16, MxK row-major) @ Bt(f16, [N][K] row-major), fused epilogues.
// Tiles staged in LDS via CDNA5 GLOBAL_LOAD_ASYNC_TO_LDS_B128 (ASYNCcnt).
#define EPI_Q   0
#define EPI_K   1
#define EPI_V   2
#define EPI_G   3
#define EPI_OUT 4

template <int EPI>
__global__ __launch_bounds__(256)
void gemm_wmma_kernel(const _Float16* __restrict__ A, const _Float16* __restrict__ Bt,
                      _Float16* __restrict__ outh, float* __restrict__ outf,
                      const float* __restrict__ bvec) {
  __shared__ _Float16 As[128][40];   // 32-K tile, +8 pad (16B-aligned rows)
  __shared__ _Float16 Bs[128][40];
  const int tid = threadIdx.x;
  const int wid = tid >> 5;
  const int m0 = blockIdx.x * 128;
  const int n0 = blockIdx.y * 128;
  const int wm = (wid & 3) * 32;     // 4 waves along M
  const int wn = (wid >> 2) * 64;    // 2 waves along N

  v8f acc[2][4];
  #pragma unroll
  for (int mi = 0; mi < 2; ++mi)
    #pragma unroll
    for (int nj = 0; nj < 4; ++nj) acc[mi][nj] = zero8();

  const int lr = tid >> 1;           // 0..127 : tile row handled by this thread
  const int lc = (tid & 1) * 16;     // 0 or 16

  for (int ks = 0; ks < 8; ++ks) {
    const int k0 = ks * 32;
    // async memory -> LDS (32B per matrix per thread), no VGPR round-trip
    async_copy_32B(&As[lr][lc], A  + (size_t)(m0 + lr) * 256 + k0 + lc);
    async_copy_32B(&Bs[lr][lc], Bt + (size_t)(n0 + lr) * 256 + k0 + lc);
    if (ks < 7) {   // pull next K-tiles toward L2/WGP$ (global_prefetch_b8)
      __builtin_prefetch(A  + (size_t)(m0 + lr) * 256 + k0 + 32 + lc, 0, 1);
      __builtin_prefetch(Bt + (size_t)(n0 + lr) * 256 + k0 + 32 + lc, 0, 1);
    }
    wait_async_all();
    __syncthreads();

    v16h af[2], bf[4];
    #pragma unroll
    for (int mi = 0; mi < 2; ++mi) af[mi] = load_frag_a(&As[wm + 16 * mi][0], 40);
    #pragma unroll
    for (int nj = 0; nj < 4; ++nj) bf[nj] = load_frag_b(&Bs[wn + 16 * nj][0], 40);
    #pragma unroll
    for (int mi = 0; mi < 2; ++mi)
      #pragma unroll
      for (int nj = 0; nj < 4; ++nj)
        acc[mi][nj] = __builtin_amdgcn_wmma_f32_16x16x32_f16(
            false, af[mi], false, bf[nj], (short)0, acc[mi][nj], false, false);
    __syncthreads();
  }

  // epilogue: acc element e, lane L -> row = e + 8*(L>>4), col = L&15 of 16x16 tile
  const int lane = tid & 31;
  const int nn = lane & 15, hh = lane >> 4;
  #pragma unroll
  for (int mi = 0; mi < 2; ++mi)
    #pragma unroll
    for (int nj = 0; nj < 4; ++nj)
      #pragma unroll
      for (int e = 0; e < 8; ++e) {
        const int grow = m0 + wm + 16 * mi + e + 8 * hh;   // 0..65535
        const int gcol = n0 + wn + 16 * nj + nn;           // 0..255
        float v = acc[mi][nj][e];
        if constexpr (EPI == EPI_Q || EPI == EPI_K) {
          if (EPI == EPI_Q) v *= 0.17677669529663687f;     // 1/sqrt(32)
          const int s = grow >> 8, r = grow & 255, h = gcol >> 5, c = gcol & 31;
          outh[(((size_t)s * N_HEAD + h) * N_RES + r) * C_H + c] = (_Float16)v;
        } else if constexpr (EPI == EPI_V) {
          const int s = grow >> 8, r = grow & 255, h = gcol >> 5, c = gcol & 31;
          outh[(((size_t)s * N_HEAD + h) * C_H + c) * N_RES + r] = (_Float16)v;  // v^T
        } else if constexpr (EPI == EPI_G) {
          v += bvec[gcol];
          v = 1.0f / (1.0f + __expf(-v));                  // sigmoid gate
          outh[(size_t)grow * 256 + gcol] = (_Float16)v;
        } else {                                           // EPI_OUT
          outf[(size_t)grow * 256 + gcol] = v + bvec[gcol];
        }
      }
}

// ---------------- kernel 5: fused attention (scores + bias + softmax + A@V + gate) ----
// One wave per (s, h, 16 query rows).
__global__ __launch_bounds__(32)
void attn_kernel(const _Float16* __restrict__ q, const _Float16* __restrict__ k,
                 const _Float16* __restrict__ vT, const float* __restrict__ bias_frag,
                 const _Float16* __restrict__ g, _Float16* __restrict__ og) {
  __shared__ _Float16 probs[16][256];
  const int qt = blockIdx.x, h = blockIdx.y, s = blockIdx.z;
  const int lane = threadIdx.x;
  const int nn = lane & 15, hh = lane >> 4;
  const size_t sh = (size_t)s * N_HEAD + h;

  // scores: 16x256 via 16 WMMA tiles, single K-step (C=32); bias added from
  // pre-swizzled C-fragment layout with two b128 loads per tile.
  v16h aq = load_frag_a(q + (sh * N_RES + qt * 16) * C_H, C_H);
  const float* bf_base = bias_frag + ((((size_t)h * 16 + qt) * 16) * 32 + lane) * 8;
  v8f sc[16];
  #pragma unroll
  for (int t = 0; t < 16; ++t) {
    v16h bk = load_frag_b(k + (sh * N_RES + t * 16) * C_H, C_H);
    sc[t] = __builtin_amdgcn_wmma_f32_16x16x32_f16(
        false, aq, false, bk, (short)0, zero8(), false, false);
    const float* bp = bf_base + (size_t)t * 256;   // t*32 lanes*8
    v4f b0 = *(const v4f*)bp;
    v4f b1 = *(const v4f*)(bp + 4);
    #pragma unroll
    for (int e = 0; e < 4; ++e) { sc[t][e] += b0[e]; sc[t][e + 4] += b1[e]; }
  }

  // in-wave softmax (wave32: each 16-lane half holds one row per acc element)
  #pragma unroll
  for (int e = 0; e < 8; ++e) {
    float mx = -3.0e38f;
    #pragma unroll
    for (int t = 0; t < 16; ++t) mx = fmaxf(mx, sc[t][e]);
    #pragma unroll
    for (int m = 8; m >= 1; m >>= 1) mx = fmaxf(mx, __shfl_xor(mx, m, 32));
    float sum = 0.0f;
    #pragma unroll
    for (int t = 0; t < 16; ++t) {
      float p = __expf(sc[t][e] - mx);
      sc[t][e] = p;
      sum += p;
    }
    #pragma unroll
    for (int m = 8; m >= 1; m >>= 1) sum += __shfl_xor(sum, m, 32);
    const float inv = 1.0f / sum;
    const int row = e + 8 * hh;
    #pragma unroll
    for (int t = 0; t < 16; ++t)
      probs[row][t * 16 + nn] = (_Float16)(sc[t][e] * inv);
  }
  __syncthreads();

  // o = probs @ v : M=16, N=32, K=256 (8 K-steps, v stored transposed [C][R])
  v8f oacc[2] = { zero8(), zero8() };
  #pragma unroll
  for (int w = 0; w < 8; ++w) {
    const int kb = w * 32;
    v16h ap = load_frag_a(&probs[0][kb], 256);
    #pragma unroll
    for (int nt = 0; nt < 2; ++nt) {
      v16h bv = load_frag_b(vT + (sh * C_H + nt * 16) * N_RES + kb, N_RES);
      oacc[nt] = __builtin_amdgcn_wmma_f32_16x16x32_f16(
          false, ap, false, bv, (short)0, oacc[nt], false, false);
    }
  }

  // gated store: og[s, r, h*32 + c] = g * o   (f16)
  #pragma unroll
  for (int nt = 0; nt < 2; ++nt)
    #pragma unroll
    for (int e = 0; e < 8; ++e) {
      const int qrow = qt * 16 + e + 8 * hh;
      const int c = nt * 16 + nn;
      const size_t idx = ((size_t)s * N_RES + qrow) * (N_HEAD * C_H) + (size_t)h * C_H + c;
      og[idx] = (_Float16)((float)g[idx] * oacc[nt][e]);
    }
}

// ---------------- workspace layout (bytes) ----------------
#define SZ_WT   ((size_t)256 * 256 * 2)            // 131072
#define SZ_ACT  ((size_t)M_ROWS * 256 * 2)         // 33554432 (f16, 65536x256)
#define SZ_BIAS ((size_t)N_HEAD * N_RES * N_RES * 4)

#define OFF_WQT  ((size_t)0)
#define OFF_WKT  (OFF_WQT + SZ_WT)
#define OFF_WVT  (OFF_WKT + SZ_WT)
#define OFF_WGT  (OFF_WVT + SZ_WT)
#define OFF_WOT  (OFF_WGT + SZ_WT)
#define OFF_MN   (OFF_WOT + SZ_WT)                 // reused as OG after projections
#define OFF_BIAS (OFF_MN + SZ_ACT)
#define OFF_Q    (OFF_BIAS + SZ_BIAS)
#define OFF_K    (OFF_Q + SZ_ACT)
#define OFF_VT   (OFF_K + SZ_ACT)
#define OFF_G    (OFF_VT + SZ_ACT)

extern "C" void kernel_launch(void* const* d_in, const int* in_sizes, int n_in,
                              void* d_out, int out_size, void* d_ws, size_t ws_size,
                              hipStream_t stream) {
  (void)in_sizes; (void)n_in; (void)out_size; (void)ws_size;
  const float* m      = (const float*)d_in[0];
  const float* z      = (const float*)d_in[1];
  const float* ln_m_w = (const float*)d_in[2];
  const float* ln_m_b = (const float*)d_in[3];
  const float* ln_z_w = (const float*)d_in[4];
  const float* ln_z_b = (const float*)d_in[5];
  const float* Wz     = (const float*)d_in[6];
  const float* Wq     = (const float*)d_in[7];
  const float* Wk     = (const float*)d_in[8];
  const float* Wv     = (const float*)d_in[9];
  const float* Wg     = (const float*)d_in[10];
  const float* bg     = (const float*)d_in[11];
  const float* Wo     = (const float*)d_in[12];
  const float* bo     = (const float*)d_in[13];
  float* out = (float*)d_out;

  char* ws = (char*)d_ws;
  _Float16* wqT  = (_Float16*)(ws + OFF_WQT);
  _Float16* wkT  = (_Float16*)(ws + OFF_WKT);
  _Float16* wvT  = (_Float16*)(ws + OFF_WVT);
  _Float16* wgT  = (_Float16*)(ws + OFF_WGT);
  _Float16* woT  = (_Float16*)(ws + OFF_WOT);
  _Float16* mn   = (_Float16*)(ws + OFF_MN);
  _Float16* og   = (_Float16*)(ws + OFF_MN);   // alias: mn dead after projections
  float*    bias = (float*)   (ws + OFF_BIAS);
  _Float16* qb   = (_Float16*)(ws + OFF_Q);
  _Float16* kb   = (_Float16*)(ws + OFF_K);
  _Float16* vT   = (_Float16*)(ws + OFF_VT);
  _Float16* gb   = (_Float16*)(ws + OFF_G);

  // 1. weights -> f16 transposed
  prep_weights_kernel<<<dim3(256, 5), 256, 0, stream>>>(Wq, Wk, Wv, Wg, Wo,
                                                        wqT, wkT, wvT, wgT, woT);
  // 2. layernorm(m) -> f16
  ln_m_kernel<<<M_ROWS, 256, 0, stream>>>(m, ln_m_w, ln_m_b, mn);
  // 3. pair bias in C-fragment layout
  pair_bias_kernel<<<dim3(N_RES, N_RES), 128, 0, stream>>>(z, ln_z_w, ln_z_b, Wz, bias);
  // 4. projections: q (scaled), k, v^T, gate
  dim3 ggrid(M_ROWS / 128, 2);
  gemm_wmma_kernel<EPI_Q><<<ggrid, 256, 0, stream>>>(mn, wqT, qb, nullptr, nullptr);
  gemm_wmma_kernel<EPI_K><<<ggrid, 256, 0, stream>>>(mn, wkT, kb, nullptr, nullptr);
  gemm_wmma_kernel<EPI_V><<<ggrid, 256, 0, stream>>>(mn, wvT, vT, nullptr, nullptr);
  gemm_wmma_kernel<EPI_G><<<ggrid, 256, 0, stream>>>(mn, wgT, gb, nullptr, bg);
  // 5. fused attention + gating -> og (overwrites mn region)
  attn_kernel<<<dim3(N_RES / 16, N_HEAD, N_SEQ), 32, 0, stream>>>(qb, kb, vT, bias, gb, og);
  // 6. output projection: out = og @ Wo + bo  (f32)
  gemm_wmma_kernel<EPI_OUT><<<ggrid, 256, 0, stream>>>(og, woT, nullptr, out, bo);
}
